// MultiHeadSelfAttention2D_34067680592127
// MI455X (gfx1250) — matrix operand order repaired and checked by reading, not compiled, not run on hardware
//
#include <hip/hip_runtime.h>

// ---------------------------------------------------------------------------
// MultiHeadSelfAttention2D for MI455X (gfx1250, wave32, WMMA 16x16x32 f16)
//
// B=8, C=128, N=4096 tokens, 4 heads, head_dim=32.  ~73 GFLOP total; working
// set ~65MB fits in 192MB L2 -> compute bound on matrix units -> f16 WMMA
// with f32 accumulation, flash-style online softmax, async double-buffered
// K/V staging (GLOBAL_LOAD_ASYNC_TO_LDS + s_wait_asynccnt).
// ---------------------------------------------------------------------------

typedef _Float16 h16;
typedef __attribute__((ext_vector_type(8)))  _Float16 v8h;
typedef __attribute__((ext_vector_type(16))) _Float16 v16h;
typedef __attribute__((ext_vector_type(8)))  float    v8f;

#define EMBED  128
#define N_TOK  4096
#define HEADS  4
#define HDIM   32

// Build a v16h WMMA fragment from two contiguous 8-half (16B) chunks.
__device__ __forceinline__ v16h frag_join(const h16* p0, const h16* p1) {
  v8h a = *(const v8h*)p0;
  v8h b = *(const v8h*)p1;
  v16h r;
#pragma unroll
  for (int i = 0; i < 8; ++i) { r[i] = a[i]; r[i + 8] = b[i]; }
  return r;
}

__device__ __forceinline__ v8f wmma16(v16h a, v16h b, v8f c) {
  // D = A(16x32) x B(32x16) + C, f32 accum
  return __builtin_amdgcn_wmma_f32_16x16x32_f16(
      false, a, false, b, (short)0, c, false, false);
}

// Async 16B copy global -> LDS (ASYNCcnt-tracked, no VGPR roundtrip).
__device__ __forceinline__ void async_cp16(h16* lds_dst, const h16* gsrc) {
  unsigned loff = (unsigned)(uintptr_t)lds_dst;  // low 32 bits = LDS offset
  asm volatile("global_load_async_to_lds_b128 %0, %1, off"
               :: "v"(loff), "v"(gsrc) : "memory");
}

// ---------------------------------------------------------------------------
// Kernel 0: convert Wq/Wk/Wv/Wo (f32, row-major [co][ci]) to f16 in workspace
// ---------------------------------------------------------------------------
__global__ __launch_bounds__(256) void k_cvt_w(const float* __restrict__ Wq,
                                               const float* __restrict__ Wk,
                                               const float* __restrict__ Wv,
                                               const float* __restrict__ Wo,
                                               h16* __restrict__ wh) {
  int idx = blockIdx.x * 256 + threadIdx.x;
  if (idx >= 4 * EMBED * EMBED) return;
  int m = idx >> 14;
  int r = idx & 16383;
  const float* s = (m == 0) ? Wq : (m == 1) ? Wk : (m == 2) ? Wv : Wo;
  wh[idx] = (h16)s[r];
}

// ---------------------------------------------------------------------------
// Kernel 1: QKV projection.  t[n][c] = x[b][c][n];  Y = t @ W^T + bias.
// Outputs: Q,K as f16 (b,h,n,d);  V transposed as f16 (b,h,d,n).
// ---------------------------------------------------------------------------
__global__ __launch_bounds__(256) void k_qkv(const float* __restrict__ x,
                                             const h16* __restrict__ wh,
                                             const float* __restrict__ bq,
                                             const float* __restrict__ bk,
                                             const float* __restrict__ bv,
                                             h16* __restrict__ Q,
                                             h16* __restrict__ K,
                                             h16* __restrict__ Vt) {
  __shared__ h16 lds_t[64][136];   // 64 tokens x 128 ch, stride 136 (16B align)

  int b  = blockIdx.x >> 6;
  int n0 = (blockIdx.x & 63) * 64;
  int tid = threadIdx.x;

  const float* xb = x + (size_t)b * EMBED * N_TOK + n0;
  for (int idx = tid; idx < 64 * EMBED; idx += 256) {
    int c = idx >> 6, n = idx & 63;
    lds_t[n][c] = (h16)xb[(size_t)c * N_TOK + n];
  }
  __syncthreads();

  int lane = tid & 31, w = tid >> 5;
  int l15 = lane & 15, hi = lane >> 4;

  // 4 row-tiles x 24 col-tiles (Q|K|V x 8) = 96 tiles; 12 per wave.
  for (int t = 0; t < 12; ++t) {
    int tile = w * 12 + t;
    int rt  = tile / 24;
    int m   = tile % 24;
    int mat = m >> 3;                // 0=Q 1=K 2=V
    int ct  = m & 7;
    int co  = ct * 16 + l15;

    const h16* wb = wh + mat * 16384 + co * 128 + 8 * hi;
    v8f acc = {};
#pragma unroll
    for (int kc = 0; kc < 4; ++kc) {
      const h16* ap = &lds_t[rt * 16 + l15][32 * kc + 8 * hi];
      v16h A = frag_join(ap, ap + 16);
      v16h B = frag_join(wb + 32 * kc, wb + 32 * kc + 16);
      acc = wmma16(A, B, acc);
    }
    float bias = (mat == 0 ? bq : mat == 1 ? bk : bv)[co];
    int h = co >> 5, d = co & 31;

    if (mat < 2) {
      h16* dst = (mat == 0 ? Q : K) + (size_t)(b * HEADS + h) * N_TOK * HDIM;
#pragma unroll
      for (int r = 0; r < 8; ++r) {
        int n = n0 + rt * 16 + r + 8 * hi;
        dst[(size_t)n * HDIM + d] = (h16)(acc[r] + bias);
      }
    } else {
      h16* dst = Vt + ((size_t)(b * HEADS + h) * HDIM + d) * N_TOK;
#pragma unroll
      for (int r = 0; r < 8; ++r) {
        int n = n0 + rt * 16 + r + 8 * hi;
        dst[n] = (h16)(acc[r] + bias);      // consecutive r -> b128 store
      }
    }
  }
}

// ---------------------------------------------------------------------------
// Kernel 2: flash attention.  Block = 8 waves x 16 queries = 128 queries of
// one (b,h).  64-key chunks, async double-buffered K/V staging in LDS,
// online softmax (max via lane shuffles, sum via P@ones WMMA).
// ---------------------------------------------------------------------------
__global__ __launch_bounds__(256) void k_attn(const h16* __restrict__ Q,
                                              const h16* __restrict__ K,
                                              const h16* __restrict__ Vt,
                                              h16* __restrict__ O) {
  __shared__ h16 ldsK[2][64][40];     // [buf][key][feat]  stride 80B
  __shared__ h16 ldsV[2][32][72];     // [buf][feat][key]  stride 144B
  __shared__ h16 ldsP[8][16][72];     // per-wave P tile [qrow][key]

  int bh   = blockIdx.x >> 5;
  int qblk = blockIdx.x & 31;
  int tid  = threadIdx.x, lane = tid & 31, w = tid >> 5;
  int l15  = lane & 15, hi = lane >> 4;

  const h16* Qb = Q  + (size_t)bh * N_TOK * HDIM;
  const h16* Kb = K  + (size_t)bh * N_TOK * HDIM;
  const h16* Vb = Vt + (size_t)bh * HDIM * N_TOK;

  int q0 = qblk * 128 + w * 16;

  // Persistent Q fragment, pre-scaled by (1/sqrt(32))*log2(e) so exp2() is
  // used directly and no per-chunk scaling is needed.
  const h16* qp = Qb + (size_t)(q0 + l15) * HDIM + 8 * hi;
  v16h Af = frag_join(qp, qp + 16);
#pragma unroll
  for (int i = 0; i < 16; ++i) Af[i] = Af[i] * (h16)0.2550300817385937f;

  // All-ones B fragment for row-sum-via-WMMA.
  v16h OnesB;
#pragma unroll
  for (int i = 0; i < 16; ++i) OnesB[i] = (h16)1.0f;

  v8f O0 = {}, O1 = {}, Lc = {};
  float M[8];
#pragma unroll
  for (int r = 0; r < 8; ++r) M[r] = -1e30f;

  // Per-thread staging assignment (2 x 16B per chunk).
  int krow = tid >> 2, kpart = tid & 3;   // K: 64 rows x 64B
  int vrow = tid >> 3, vpart = tid & 7;   // V: 32 rows x 128B

  // Prologue: issue chunk 0 into buffer 0.
  async_cp16(&ldsK[0][krow][kpart * 8],
             &Kb[(size_t)krow * HDIM + kpart * 8]);
  async_cp16(&ldsV[0][vrow][vpart * 8],
             &Vb[(size_t)vrow * N_TOK + vpart * 8]);

  for (int kc = 0; kc < 64; ++kc) {
    int cur = kc & 1;
    __syncthreads();   // everyone finished reading buf[cur^1] (chunk kc-1)
    if (kc < 63) {
      int key0 = (kc + 1) * 64;
      async_cp16(&ldsK[cur ^ 1][krow][kpart * 8],
                 &Kb[(size_t)(key0 + krow) * HDIM + kpart * 8]);
      async_cp16(&ldsV[cur ^ 1][vrow][vpart * 8],
                 &Vb[(size_t)vrow * N_TOK + key0 + vpart * 8]);
      asm volatile("s_wait_asynccnt 2" ::: "memory");  // chunk kc arrived
    } else {
      asm volatile("s_wait_asynccnt 0" ::: "memory");
    }
    __syncthreads();   // chunk kc visible to all waves

    // S = Q K^T for 4 sub-tiles of 16 keys (already in log2/scale domain)
    v8f S[4];
#pragma unroll
    for (int kt = 0; kt < 4; ++kt) {
      const h16* kp = &ldsK[cur][kt * 16 + l15][8 * hi];
      v16h Bf = frag_join(kp, kp + 16);
      v8f z = {};
      S[kt] = wmma16(Af, Bf, z);
    }

    // Online softmax per query row (row = r + 8*hi; 16 lanes hold 16 keys)
#pragma unroll
    for (int r = 0; r < 8; ++r) {
      float s0 = S[0][r], s1 = S[1][r], s2 = S[2][r], s3 = S[3][r];
      float mx = fmaxf(fmaxf(s0, s1), fmaxf(s2, s3));
#pragma unroll
      for (int msk = 1; msk < 16; msk <<= 1)
        mx = fmaxf(mx, __shfl_xor(mx, msk, 32));
      float Mn = fmaxf(M[r], mx);
      float alpha = __builtin_amdgcn_exp2f(M[r] - Mn);
      M[r] = Mn;
      float p0 = __builtin_amdgcn_exp2f(s0 - Mn);
      float p1 = __builtin_amdgcn_exp2f(s1 - Mn);
      float p2 = __builtin_amdgcn_exp2f(s2 - Mn);
      float p3 = __builtin_amdgcn_exp2f(s3 - Mn);
      O0[r] *= alpha;
      O1[r] *= alpha;
      Lc[r] *= alpha;
      int row = r + 8 * hi;
      ldsP[w][row][l15]      = (h16)p0;
      ldsP[w][row][l15 + 16] = (h16)p1;
      ldsP[w][row][l15 + 32] = (h16)p2;
      ldsP[w][row][l15 + 48] = (h16)p3;
    }

    // O += P(16x64) @ V(64x32);  L += P @ ones (row sums, same C layout)
#pragma unroll
    for (int af = 0; af < 2; ++af) {
      const h16* pp = &ldsP[w][l15][32 * af + 16 * hi];
      v16h Ap = frag_join(pp, pp + 8);
      const h16* v0 = &ldsV[cur][l15][32 * af + 16 * hi];
      const h16* v1 = &ldsV[cur][16 + l15][32 * af + 16 * hi];
      v16h B0 = frag_join(v0, v0 + 8);
      v16h B1 = frag_join(v1, v1 + 8);
      O0 = wmma16(Ap, B0, O0);
      O1 = wmma16(Ap, B1, O1);
      Lc = wmma16(Ap, OnesB, Lc);
    }
  }

  // Normalize and store O as f16 (b,h,n,d)
  h16* Ob = O + (size_t)bh * N_TOK * HDIM;
#pragma unroll
  for (int r = 0; r < 8; ++r) {
    float inv = 1.0f / Lc[r];
    int n = q0 + r + 8 * hi;
    Ob[(size_t)n * HDIM + l15]      = (h16)(O0[r] * inv);
    Ob[(size_t)n * HDIM + l15 + 16] = (h16)(O1[r] * inv);
  }
}

// ---------------------------------------------------------------------------
// Kernel 3: output projection.  Ocat[n][c=h*32+d] @ Wo^T + bo -> out[b][c][n]
// ---------------------------------------------------------------------------
__global__ __launch_bounds__(256) void k_proj(const h16* __restrict__ O,
                                              const h16* __restrict__ wo,
                                              const float* __restrict__ bo,
                                              float* __restrict__ out) {
  int b  = blockIdx.x >> 6;
  int n0 = (blockIdx.x & 63) * 64;
  int tid = threadIdx.x, lane = tid & 31, w = tid >> 5;
  int l15 = lane & 15, hi = lane >> 4;

  for (int t = 0; t < 4; ++t) {
    int tile = w * 4 + t;
    int rt = tile >> 3, ct = tile & 7;
    int co = ct * 16 + l15;

    const h16* wb = wo + co * 128 + 8 * hi;
    v8f acc = {};
#pragma unroll
    for (int kc = 0; kc < 4; ++kc) {     // kc == head index (32 feats each)
      const h16* ap =
          O + ((size_t)(b * HEADS + kc) * N_TOK + n0 + rt * 16 + l15) * HDIM +
          8 * hi;
      v16h A = frag_join(ap, ap + 16);
      v16h B = frag_join(wb + 32 * kc, wb + 32 * kc + 16);
      acc = wmma16(A, B, acc);
    }
    float bias = bo[co];
    float* dst = out + ((size_t)b * EMBED + co) * N_TOK;
#pragma unroll
    for (int r = 0; r < 8; ++r) {
      dst[n0 + rt * 16 + r + 8 * hi] = acc[r] + bias;
    }
  }
}

// ---------------------------------------------------------------------------
// Launch
// ---------------------------------------------------------------------------
extern "C" void kernel_launch(void* const* d_in, const int* in_sizes, int n_in,
                              void* d_out, int out_size, void* d_ws,
                              size_t ws_size, hipStream_t stream) {
  (void)in_sizes; (void)n_in; (void)out_size; (void)ws_size;

  const float* x  = (const float*)d_in[0];
  const float* Wq = (const float*)d_in[1];
  const float* bq = (const float*)d_in[2];
  const float* Wk = (const float*)d_in[3];
  const float* bk = (const float*)d_in[4];
  const float* Wv = (const float*)d_in[5];
  const float* bv = (const float*)d_in[6];
  const float* Wo = (const float*)d_in[7];
  const float* bo = (const float*)d_in[8];
  float* out = (float*)d_out;

  // Workspace layout (half elements):
  //   [0, 65536)       : Wq,Wk,Wv,Wo in f16
  //   + 4 x 4194304    : Q, K, Vt, O  (f16)   -> ~33.7 MB total
  h16* wh = (h16*)d_ws;
  h16* Qh = wh + 65536;
  h16* Kh = Qh + 4194304;
  h16* Vh = Kh + 4194304;
  h16* Oh = Vh + 4194304;

  k_cvt_w<<<256, 256, 0, stream>>>(Wq, Wk, Wv, Wo, wh);
  k_qkv  <<<512, 256, 0, stream>>>(x, wh, bq, bk, bv, Qh, Kh, Vh);
  k_attn <<<1024, 256, 0, stream>>>(Qh, Kh, Vh, Oh);
  k_proj <<<512, 256, 0, stream>>>(Oh, wh + 3 * 16384, bo, out);
}
